// MoELayer_83837761618193
// MI455X (gfx1250) — matrix operand order repaired and checked by reading, not compiled, not run on hardware
//
#include <hip/hip_runtime.h>
#include <hip/hip_bf16.h>
#include <math.h>

// ---------------------------------------------------------------------------
// MoE layer for MI455X (gfx1250), wave32 + WMMA bf16 + async global->LDS DMA.
//   B=4 T=2048 D=1024 F=4096 E=8 TOP_K=2, N = B*T = 8192 tokens.
// Workspace layout (requires ~288 MB):
//   [0..)      counts[8]           (int)
//   [64..)     hoff[9]             (int, exclusive scan of counts)
//   [128..)    probs_sum[8]        (float)
//   [4096..)   blk_probs[1024*8]   (float)
//   [64KB..)   tlist[8*8192]       (int)
//   [320KB..)  cwlist[8*8192]      (float)
//   [1MB..)    xb[8192*1024]       (bf16, x converted)
//   [32MB..)   w1t[8][4096][1024]  (bf16, w1 transposed per expert)
//   [96MB..)   w2t[8][1024][4096]  (bf16, w2 transposed per expert)
//   [160MB..)  hbuf[16384*4096]    (bf16, gathered hidden activations)
// ---------------------------------------------------------------------------

#define MOE_N 8192
#define MOE_D 1024
#define MOE_F 4096
#define MOE_E 8
#define KT 64        // k-depth per LDS buffer
#define KPAD 72      // padded row length in halfwords (144B: 16B-aligned, bank-conflict-free)

typedef __attribute__((ext_vector_type(16))) __bf16 bf16x16;
typedef __attribute__((ext_vector_type(8)))  float  floatx8;

__device__ __forceinline__ unsigned short f2bf(float f) {
  return __builtin_bit_cast(unsigned short, static_cast<__bf16>(f));
}
__device__ __forceinline__ unsigned pack2(float lo, float hi) {
  return (unsigned)f2bf(lo) | ((unsigned)f2bf(hi) << 16);
}
__device__ __forceinline__ bf16x16 ld_frag(const __bf16* lo, const __bf16* hi) {
  union { bf16x16 v; uint4 q[2]; } u;
  u.q[0] = *reinterpret_cast<const uint4*>(lo);
  u.q[1] = *reinterpret_cast<const uint4*>(hi);
  return u.v;
}
__device__ __forceinline__ float gelu_exact(float v) {
  return 0.5f * v * (1.0f + erff(v * 0.7071067811865475f));
}
// Low 32 bits of a generic pointer into LDS == LDS byte offset (AS3 aperture).
__device__ __forceinline__ unsigned lds_off(const void* p) {
  return (unsigned)(unsigned long long)(size_t)p;
}
// Async DMA: copy 16 bytes global -> LDS, tracked by ASYNCcnt.
__device__ __forceinline__ void async_b128(unsigned lds_addr, unsigned long long gaddr) {
  asm volatile("global_load_async_to_lds_b128 %0, %1, off"
               :
               : "v"(lds_addr), "v"(gaddr)
               : "memory");
}
__device__ __forceinline__ void wait_async0() {
  asm volatile("s_wait_asynccnt 0x0" ::: "memory");
}

// ---------------------------------------------------------------------------
// init: zero output and per-expert counters
// ---------------------------------------------------------------------------
__global__ void moe_init(float* __restrict__ out, int n, int* __restrict__ counts) {
  int gid = blockIdx.x * blockDim.x + threadIdx.x;
  for (int i = gid; i < n; i += gridDim.x * blockDim.x) out[i] = 0.0f;
  if (gid < MOE_E) counts[gid] = 0;
}

// ---------------------------------------------------------------------------
// cvt_x: fp32 -> bf16, row-major (8 elements / thread)
// ---------------------------------------------------------------------------
__global__ void moe_cvt_x(const float* __restrict__ in, __bf16* __restrict__ out,
                          int n8) {
  int i = blockIdx.x * blockDim.x + threadIdx.x;
  if (i < n8) {
    const float4* p = reinterpret_cast<const float4*>(in + (size_t)i * 8);
    float4 a = p[0], b = p[1];
    uint4 q = make_uint4(pack2(a.x, a.y), pack2(a.z, a.w),
                         pack2(b.x, b.y), pack2(b.z, b.w));
    *reinterpret_cast<uint4*>(out + (size_t)i * 8) = q;
  }
}

// ---------------------------------------------------------------------------
// transpose: in fp32 [E][R][C] -> out bf16 [E][C][R]; LDS-tiled (32x32)
// ---------------------------------------------------------------------------
__global__ __launch_bounds__(256) void moe_transpose_bf16(
    const float* __restrict__ in, __bf16* __restrict__ outp, int R, int C) {
  __shared__ float tile[32][33];
  const float* ine = in + (size_t)blockIdx.z * R * C;
  __bf16* oute = outp + (size_t)blockIdx.z * R * C;
  const int tx = threadIdx.x, ty = threadIdx.y;  // (32, 8)
  const int c0 = blockIdx.x * 32, r0 = blockIdx.y * 32;
#pragma unroll
  for (int j = 0; j < 4; ++j)
    tile[ty + j * 8][tx] = ine[(size_t)(r0 + ty + j * 8) * C + c0 + tx];
  __syncthreads();
#pragma unroll
  for (int j = 0; j < 4; ++j)
    oute[(size_t)(c0 + ty + j * 8) * R + r0 + tx] =
        static_cast<__bf16>(tile[tx][ty + j * 8]);
}

// ---------------------------------------------------------------------------
// router: one wave per token. logits -> softmax -> top2 -> expert lists.
// ---------------------------------------------------------------------------
__global__ __launch_bounds__(256) void moe_router(
    const float* __restrict__ x, const float* __restrict__ gate_w,
    int* __restrict__ counts, int* __restrict__ tlist,
    float* __restrict__ cwlist, float* __restrict__ blk_probs) {
  __shared__ float wave_p[8][MOE_E];
  const int t = threadIdx.x;
  const int wid = t >> 5, lane = t & 31;
  const int tk = blockIdx.x * 8 + wid;

  float acc[MOE_E];
#pragma unroll
  for (int e = 0; e < MOE_E; ++e) acc[e] = 0.0f;

  const float* xr = x + (size_t)tk * MOE_D;
  for (int d = lane; d < MOE_D; d += 32) {
    float xv = xr[d];
    const float4* g = reinterpret_cast<const float4*>(gate_w + (size_t)d * MOE_E);
    float4 g0 = g[0], g1 = g[1];
    acc[0] += xv * g0.x; acc[1] += xv * g0.y; acc[2] += xv * g0.z; acc[3] += xv * g0.w;
    acc[4] += xv * g1.x; acc[5] += xv * g1.y; acc[6] += xv * g1.z; acc[7] += xv * g1.w;
  }
#pragma unroll
  for (int off = 16; off > 0; off >>= 1)
#pragma unroll
    for (int e = 0; e < MOE_E; ++e) acc[e] += __shfl_xor(acc[e], off, 32);

  if (lane == 0) {
    float mx = acc[0];
#pragma unroll
    for (int e = 1; e < MOE_E; ++e) mx = fmaxf(mx, acc[e]);
    float p[MOE_E], s = 0.0f;
#pragma unroll
    for (int e = 0; e < MOE_E; ++e) { p[e] = expf(acc[e] - mx); s += p[e]; }
    float inv = 1.0f / s;
#pragma unroll
    for (int e = 0; e < MOE_E; ++e) { p[e] *= inv; wave_p[wid][e] = p[e]; }
    int e1 = 0;
#pragma unroll
    for (int e = 1; e < MOE_E; ++e) if (p[e] > p[e1]) e1 = e;
    int e2 = (e1 == 0) ? 1 : 0;
#pragma unroll
    for (int e = 0; e < MOE_E; ++e) if (e != e1 && p[e] > p[e2]) e2 = e;
    float wsum = p[e1] + p[e2] + 1e-9f;
    float c1 = p[e1] / wsum, c2 = p[e2] / wsum;
    int s1 = atomicAdd(&counts[e1], 1);
    tlist[e1 * MOE_N + s1] = tk; cwlist[e1 * MOE_N + s1] = c1;
    int s2 = atomicAdd(&counts[e2], 1);
    tlist[e2 * MOE_N + s2] = tk; cwlist[e2 * MOE_N + s2] = c2;
  }
  __syncthreads();
  if (t < MOE_E) {
    float s = 0.0f;
#pragma unroll
    for (int w = 0; w < 8; ++w) s += wave_p[w][t];
    blk_probs[blockIdx.x * MOE_E + t] = s;
  }
}

__global__ void moe_scan(const int* __restrict__ counts, int* __restrict__ hoff) {
  if (threadIdx.x == 0 && blockIdx.x == 0) {
    int s = 0;
    for (int e = 0; e < MOE_E; ++e) { hoff[e] = s; s += counts[e]; }
    hoff[MOE_E] = s;
  }
}

// ---------------------------------------------------------------------------
// ffn1: h = gelu(Xg @ w1[e] + b1[e]) -> hbuf (bf16).
// 128x128 tile, K = D = 1024, 64-deep k buffers (one barrier per 16 WMMA/wave).
// All tile loads are async global->LDS b128 DMA (operands pre-converted to
// bf16 in the exact LDS fragment layout).
// ---------------------------------------------------------------------------
__global__ __launch_bounds__(256) void moe_ffn1(
    const __bf16* __restrict__ xb, const __bf16* __restrict__ w1t,
    const float* __restrict__ b1, const int* __restrict__ tlist,
    const int* __restrict__ counts, const int* __restrict__ hoff,
    __bf16* __restrict__ hbuf) {
  const int e = blockIdx.z;
  const int cnt = counts[e];
  const int m0 = blockIdx.y * 128;
  if (m0 >= cnt) return;
  const int f0 = blockIdx.x * 128;
  const int hb = hoff[e];
  const __bf16* w1e = w1t + (size_t)e * MOE_F * MOE_D;  // [F][D]
  const float* b1e = b1 + (size_t)e * MOE_F;

  __shared__ __align__(16) __bf16 As[2][128][KPAD];  // [m][k], rows padded
  __shared__ __align__(16) __bf16 Bs[2][128][KPAD];  // [n][k], rows padded
  __shared__ int tok_s[128];

  const int t = threadIdx.x;
  if (t < 128) {
    int slot = m0 + t;
    tok_s[t] = (slot < cnt) ? tlist[e * MOE_N + slot] : -1;
  }
  __syncthreads();

  const int row = t >> 1, seg = t & 1;  // 2 threads/row, 32 halfs (64B) each
  int tokn = tok_s[row];
  if (tokn < 0) tokn = tok_s[0];  // clamp: garbage only lands in discarded rows
  const unsigned long long a_base =
      (unsigned long long)(size_t)(xb + (size_t)tokn * MOE_D + seg * 32);
  const unsigned long long b_base =
      (unsigned long long)(size_t)(w1e + (size_t)(f0 + row) * MOE_D + seg * 32);
  unsigned alA[2], alB[2];
#pragma unroll
  for (int b = 0; b < 2; ++b) {
    alA[b] = lds_off(&As[b][row][seg * 32]);
    alB[b] = lds_off(&Bs[b][row][seg * 32]);
  }

  auto issue = [&](int kk, int buf) {
    unsigned long long ka = a_base + (unsigned long long)kk * (KT * 2);
    unsigned long long kb = b_base + (unsigned long long)kk * (KT * 2);
#pragma unroll
    for (int i = 0; i < 4; ++i) async_b128(alA[buf] + 16 * i, ka + 16ull * i);
#pragma unroll
    for (int i = 0; i < 4; ++i) async_b128(alB[buf] + 16 * i, kb + 16ull * i);
  };

  const int wid = t >> 5, lane = t & 31;
  const int m_w = (wid >> 1) * 32, n_w = (wid & 1) * 64;
  const int lrow = lane & 15, khalf = lane >> 4;

  floatx8 acc[2][4];
#pragma unroll
  for (int mt = 0; mt < 2; ++mt)
#pragma unroll
    for (int nt = 0; nt < 4; ++nt)
      acc[mt][nt] = floatx8{0.f, 0.f, 0.f, 0.f, 0.f, 0.f, 0.f, 0.f};

  issue(0, 0);
  wait_async0();
  __syncthreads();
  const int NK = MOE_D / KT;  // 16
  for (int kk = 0; kk < NK; ++kk) {
    const int buf = kk & 1;
    if (kk + 1 < NK) issue(kk + 1, buf ^ 1);
#pragma unroll
    for (int sub = 0; sub < 2; ++sub) {
      const int ko = sub * 32;
      bf16x16 afr[2], bfr[4];
#pragma unroll
      for (int mt = 0; mt < 2; ++mt) {
        const __bf16* r = &As[buf][m_w + mt * 16 + lrow][ko];
        afr[mt] = ld_frag(r + khalf * 8, r + 16 + khalf * 8);
      }
#pragma unroll
      for (int nt = 0; nt < 4; ++nt) {
        const __bf16* r = &Bs[buf][n_w + nt * 16 + lrow][ko];
        bfr[nt] = ld_frag(r + khalf * 8, r + 16 + khalf * 8);
      }
#pragma unroll
      for (int mt = 0; mt < 2; ++mt)
#pragma unroll
        for (int nt = 0; nt < 4; ++nt)
          acc[mt][nt] = __builtin_amdgcn_wmma_f32_16x16x32_bf16(
              false, afr[mt], false, bfr[nt], (short)0, acc[mt][nt], false, false);
    }
    wait_async0();
    __syncthreads();
  }

#pragma unroll
  for (int mt = 0; mt < 2; ++mt)
#pragma unroll
    for (int nt = 0; nt < 4; ++nt)
#pragma unroll
      for (int r = 0; r < 8; ++r) {
        int M = m_w + mt * 16 + khalf * 8 + r;
        int slot = m0 + M;
        if (slot < cnt) {
          int f = f0 + n_w + nt * 16 + lrow;
          float v = acc[mt][nt][r] + b1e[f];
          v = gelu_exact(v);
          hbuf[(size_t)(hb + slot) * MOE_F + f] = static_cast<__bf16>(v);
        }
      }
}

// ---------------------------------------------------------------------------
// ffn2: out[tok] += cw * (h @ w2[e] + b2[e]); K = F = 4096. Same async engine.
// ---------------------------------------------------------------------------
__global__ __launch_bounds__(256) void moe_ffn2(
    const __bf16* __restrict__ hbuf, const __bf16* __restrict__ w2t,
    const float* __restrict__ b2, const int* __restrict__ tlist,
    const float* __restrict__ cwlist, const int* __restrict__ counts,
    const int* __restrict__ hoff, float* __restrict__ out) {
  const int e = blockIdx.z;
  const int cnt = counts[e];
  const int m0 = blockIdx.y * 128;
  if (m0 >= cnt) return;
  const int d0 = blockIdx.x * 128;
  const int hb = hoff[e];
  const __bf16* w2e = w2t + (size_t)e * MOE_D * MOE_F;  // [D][F]
  const float* b2e = b2 + (size_t)e * MOE_D;

  __shared__ __align__(16) __bf16 As[2][128][KPAD];
  __shared__ __align__(16) __bf16 Bs[2][128][KPAD];
  __shared__ int tok_s[128];
  __shared__ float cw_s[128];

  const int t = threadIdx.x;
  if (t < 128) {
    int slot = m0 + t;
    if (slot < cnt) {
      tok_s[t] = tlist[e * MOE_N + slot];
      cw_s[t] = cwlist[e * MOE_N + slot];
    } else {
      tok_s[t] = -1;
      cw_s[t] = 0.0f;
    }
  }
  __syncthreads();

  const int row = t >> 1, seg = t & 1;
  int slotc = m0 + row;
  if (slotc >= cnt) slotc = cnt - 1;  // clamp; results for padded rows discarded
  const unsigned long long a_base = (unsigned long long)(size_t)(
      hbuf + (size_t)(hb + slotc) * MOE_F + seg * 32);
  const unsigned long long b_base = (unsigned long long)(size_t)(
      w2e + (size_t)(d0 + row) * MOE_F + seg * 32);
  unsigned alA[2], alB[2];
#pragma unroll
  for (int b = 0; b < 2; ++b) {
    alA[b] = lds_off(&As[b][row][seg * 32]);
    alB[b] = lds_off(&Bs[b][row][seg * 32]);
  }

  auto issue = [&](int kk, int buf) {
    unsigned long long ka = a_base + (unsigned long long)kk * (KT * 2);
    unsigned long long kb = b_base + (unsigned long long)kk * (KT * 2);
#pragma unroll
    for (int i = 0; i < 4; ++i) async_b128(alA[buf] + 16 * i, ka + 16ull * i);
#pragma unroll
    for (int i = 0; i < 4; ++i) async_b128(alB[buf] + 16 * i, kb + 16ull * i);
  };

  const int wid = t >> 5, lane = t & 31;
  const int m_w = (wid >> 1) * 32, n_w = (wid & 1) * 64;
  const int lrow = lane & 15, khalf = lane >> 4;

  floatx8 acc[2][4];
#pragma unroll
  for (int mt = 0; mt < 2; ++mt)
#pragma unroll
    for (int nt = 0; nt < 4; ++nt)
      acc[mt][nt] = floatx8{0.f, 0.f, 0.f, 0.f, 0.f, 0.f, 0.f, 0.f};

  issue(0, 0);
  wait_async0();
  __syncthreads();
  const int NK = MOE_F / KT;  // 64
  for (int kk = 0; kk < NK; ++kk) {
    const int buf = kk & 1;
    if (kk + 1 < NK) issue(kk + 1, buf ^ 1);
#pragma unroll
    for (int sub = 0; sub < 2; ++sub) {
      const int ko = sub * 32;
      bf16x16 afr[2], bfr[4];
#pragma unroll
      for (int mt = 0; mt < 2; ++mt) {
        const __bf16* r = &As[buf][m_w + mt * 16 + lrow][ko];
        afr[mt] = ld_frag(r + khalf * 8, r + 16 + khalf * 8);
      }
#pragma unroll
      for (int nt = 0; nt < 4; ++nt) {
        const __bf16* r = &Bs[buf][n_w + nt * 16 + lrow][ko];
        bfr[nt] = ld_frag(r + khalf * 8, r + 16 + khalf * 8);
      }
#pragma unroll
      for (int mt = 0; mt < 2; ++mt)
#pragma unroll
        for (int nt = 0; nt < 4; ++nt)
          acc[mt][nt] = __builtin_amdgcn_wmma_f32_16x16x32_bf16(
              false, afr[mt], false, bfr[nt], (short)0, acc[mt][nt], false, false);
    }
    wait_async0();
    __syncthreads();
  }

#pragma unroll
  for (int mt = 0; mt < 2; ++mt)
#pragma unroll
    for (int nt = 0; nt < 4; ++nt)
#pragma unroll
      for (int r = 0; r < 8; ++r) {
        int M = m_w + mt * 16 + khalf * 8 + r;
        int slot = m0 + M;
        if (slot < cnt) {
          int d = d0 + n_w + nt * 16 + lrow;
          float v = acc[mt][nt][r] + b2e[d];
          unsafeAtomicAdd(&out[(size_t)tok_s[M] * MOE_D + d], cw_s[M] * v);
        }
      }
}

// ---------------------------------------------------------------------------
__global__ void moe_reduce_probs(const float* __restrict__ blk_probs,
                                 float* __restrict__ probs_sum) {
  int t = threadIdx.x;
  if (t < MOE_E) {
    float s = 0.0f;
    for (int b = 0; b < MOE_N / 8; ++b) s += blk_probs[b * MOE_E + t];
    probs_sum[t] = s;
  }
}

__global__ void moe_aux(const float* __restrict__ probs_sum,
                        float* __restrict__ out_last) {
  if (threadIdx.x == 0 && blockIdx.x == 0) {
    float a = 0.0f;
    for (int e = 0; e < MOE_E; ++e) {
      float v = probs_sum[e] / (float)MOE_N - 1.0f / (float)MOE_E;
      a += v * v;
    }
    *out_last = a;
  }
}

// ---------------------------------------------------------------------------
extern "C" void kernel_launch(void* const* d_in, const int* in_sizes, int n_in,
                              void* d_out, int out_size, void* d_ws, size_t ws_size,
                              hipStream_t stream) {
  const float* x = (const float*)d_in[0];
  const float* gate_w = (const float*)d_in[1];
  const float* w1 = (const float*)d_in[2];
  const float* b1 = (const float*)d_in[3];
  const float* w2 = (const float*)d_in[4];
  const float* b2 = (const float*)d_in[5];
  float* out = (float*)d_out;

  char* ws = (char*)d_ws;
  int* counts = (int*)(ws + 0);
  int* hoff = (int*)(ws + 64);
  float* probs_sum = (float*)(ws + 128);
  float* blk_probs = (float*)(ws + 4096);
  int* tlist = (int*)(ws + 65536);
  float* cwlist = (float*)(ws + 65536 + MOE_E * MOE_N * sizeof(int));
  __bf16* xb = (__bf16*)(ws + (1ull << 20));
  __bf16* w1t = (__bf16*)(ws + (32ull << 20));
  __bf16* w2t = (__bf16*)(ws + (96ull << 20));
  __bf16* hbuf = (__bf16*)(ws + (160ull << 20));

  moe_init<<<4096, 256, 0, stream>>>(out, out_size, counts);
  moe_cvt_x<<<(MOE_N * MOE_D / 8 + 255) / 256, 256, 0, stream>>>(
      x, xb, MOE_N * MOE_D / 8);
  {
    dim3 blk(32, 8);
    dim3 g1(MOE_F / 32, MOE_D / 32, MOE_E);  // w1: [D][F] -> [F][D]
    moe_transpose_bf16<<<g1, blk, 0, stream>>>(w1, w1t, MOE_D, MOE_F);
    dim3 g2(MOE_D / 32, MOE_F / 32, MOE_E);  // w2: [F][D] -> [D][F]
    moe_transpose_bf16<<<g2, blk, 0, stream>>>(w2, w2t, MOE_F, MOE_D);
  }
  moe_router<<<MOE_N / 8, 256, 0, stream>>>(x, gate_w, counts, tlist, cwlist,
                                            blk_probs);
  moe_scan<<<1, 1, 0, stream>>>(counts, hoff);
  moe_reduce_probs<<<1, 32, 0, stream>>>(blk_probs, probs_sum);

  dim3 ga(MOE_F / 128, MOE_N / 128, MOE_E);  // 32 x 64 x 8
  moe_ffn1<<<ga, 256, 0, stream>>>(xb, w1t, b1, tlist, counts, hoff, hbuf);

  dim3 gb(MOE_D / 128, MOE_N / 128, MOE_E);  // 8 x 64 x 8
  moe_ffn2<<<gb, 256, 0, stream>>>(hbuf, w2t, b2, tlist, cwlist, counts, hoff,
                                   out);

  moe_aux<<<1, 1, 0, stream>>>(probs_sum, out + (size_t)MOE_N * MOE_D);
}
